// GECOR_70153995813114
// MI455X (gfx1250) — matrix operand order repaired and checked by reading, not compiled, not run on hardware
//
#include <hip/hip_runtime.h>
#include <math.h>

// CDNA5 / gfx1250: wave32, WMMA 16x16x32 f16 -> f32 accumulate.
typedef __attribute__((ext_vector_type(16))) _Float16 v16h;
typedef __attribute__((ext_vector_type(8)))  _Float16 v8h;
typedef __attribute__((ext_vector_type(8)))  float    v8f;

#define B_   8
#define H_   512
#define V_   32000
#define TH_  512
#define TU_  128
#define VT_  32512      /* V_ + TH_ */
#define EPS_ 1e-10f

// ---------------- WMMA fragment loaders (ISA 7.12.2 layouts, wave32) ----------------
// A tile stored [m][k] row-major (ldk halves per row). Lane holds row m = lane%16,
// wave halves split K: lanes 0-15 take K {0..7, 16..23}, lanes 16-31 take +8.
// Both 8-half chunks are 16B aligned -> ds_load_b128 pairs.
__device__ __forceinline__ v16h ld_afrag(const _Float16* A, int lane, int ldk) {
  int m = lane & 15, lh = lane >> 4;
  const _Float16* p = A + m * ldk + lh * 8;
  v8h lo = *(const v8h*)(p);
  v8h hi = *(const v8h*)(p + 16);
  return __builtin_shufflevector(lo, hi, 0,1,2,3,4,5,6,7,8,9,10,11,12,13,14,15);
}
// B tile stored K-major: Wt[k*ldn + n]; lane column n = lane%16, elements walk K
// like A: lanes 0-15 take K {0..7, 16..23}, lanes 16-31 take +8 (strided gather).
__device__ __forceinline__ v16h ld_bfrag(const _Float16* Wt, int lane, int ldn) {
  int n = lane & 15, lh = lane >> 4;
  v16h b;
#pragma unroll
  for (int s = 0; s < 16; ++s) {
    int k = (s < 8) ? (lh * 8 + s) : (16 + lh * 8 + (s - 8));
    b[s] = Wt[k * ldn + n];
  }
  return b;
}

// ---------------- utility ----------------
__global__ void zero_kernel(float* p, int n) {
  int i = blockIdx.x * blockDim.x + threadIdx.x;
  if (i < n) p[i] = 0.f;
}

// A1 = h @ attn_W[:, :H].T + attn_b ; gh = h @ gru_Whh.T + gru_bhh ; c_embed gather
__global__ void prep_kernel(const float* __restrict__ h, const float* __restrict__ attnW,
                            const float* __restrict__ attnb,
                            const float* __restrict__ Whh, const float* __restrict__ bhh,
                            const float* __restrict__ embW, const int* __restrict__ ct,
                            float* __restrict__ A1, float* __restrict__ gh,
                            float* __restrict__ cemb) {
  int idx = blockIdx.x * blockDim.x + threadIdx.x;
  if (idx >= B_ * 2560) return;
  int b = idx / 2560, r = idx % 2560;
  const float* hb = h + b * H_;
  if (r < 512) {
    float acc = attnb[r];
    const float* wrow = attnW + (size_t)r * 1024;   // first H cols of attn_W row r
    for (int k = 0; k < 512; ++k) acc += hb[k] * wrow[k];
    A1[b * 512 + r] = acc;
  } else if (r < 2048) {
    int j = r - 512;
    float acc = bhh[j];
    const float* wrow = Whh + (size_t)j * 512;
    for (int k = 0; k < 512; ++k) acc += hb[k] * wrow[k];
    gh[b * 1536 + j] = acc;
  } else {
    int j = r - 2048;
    cemb[b * 512 + j] = embW[(size_t)ct[b] * 512 + j];
  }
}

// Shared WMMA GEMM+reduce kernel for the two [Mx512]x[512x512] GEMMs.
// mode 0 (energy): out[b,t] += sum_n attn_v[n] * tanh(C + A1[b,n])
// mode 1 (copy)  : out[b,t] += sum_n tanh(C + copy2_b[n]) * h_new[b,n]
// Rows r of Aglob map to (t = r/8, b = r%8) matching [T,B,H] memory layout.
__global__ void gemm_reduce_kernel(const float* __restrict__ Aglob,
                                   const float* __restrict__ W,
                                   int wStride, int wOff,
                                   const float* __restrict__ bias1,
                                   const float* __restrict__ vec,
                                   float* __restrict__ outAcc,
                                   int Mrows, int mTilesPerBlock, int outStride, int mode) {
  extern __shared__ _Float16 lds[];
  _Float16* sW = lds;            // [512 k][64 n] K-major f16 weight slice
  _Float16* sA = lds + 64 * 512; // [16 m][512 k] f16 activation tile
  int tid = threadIdx.x;         // 128 threads = 4 waves
  int wave = tid >> 5, lane = tid & 31;
  int n0b = blockIdx.x * 64;

  // stage weights transposed to K-major (global reads stay k-contiguous/coalesced)
  for (int i = tid; i < 64 * 512; i += 128) {
    int n = i >> 9, k = i & 511;
    sW[k * 64 + n] = (_Float16)W[(size_t)(n0b + n) * wStride + wOff + k];
  }
  __syncthreads();

  for (int mt = 0; mt < mTilesPerBlock; ++mt) {
    int m0 = (blockIdx.y * mTilesPerBlock + mt) * 16;
    for (int i = tid; i < 16 * 512; i += 128) {
      int r = i >> 9, k = i & 511;
      int row = m0 + r;
      sA[i] = (row < Mrows) ? (_Float16)Aglob[(size_t)row * 512 + k] : (_Float16)0.f;
    }
    __syncthreads();

    v8f c = {0.f, 0.f, 0.f, 0.f, 0.f, 0.f, 0.f, 0.f};
    int n0w = wave * 16;
#pragma unroll 4
    for (int kk = 0; kk < 16; ++kk) {
      v16h a = ld_afrag(sA + kk * 32, lane, 512);
      v16h b = ld_bfrag(sW + kk * 32 * 64 + n0w, lane, 64);
      c = __builtin_amdgcn_wmma_f32_16x16x32_f16(false, a, false, b, (short)0, c, false, false);
    }

    int nl = lane & 15, half = lane >> 4;
    int gn = n0b + n0w + nl;
#pragma unroll
    for (int r = 0; r < 8; ++r) {
      int m = r + 8 * half;
      int row = m0 + m;
      int bb = row & 7, tt = row >> 3;
      float v;
      if (mode == 0) v = vec[gn] * tanhf(c[r] + bias1[bb * 512 + gn]);
      else           v = tanhf(c[r] + bias1[gn]) * vec[bb * 512 + gn];
      v += __shfl_xor(v, 1);
      v += __shfl_xor(v, 2);
      v += __shfl_xor(v, 4);
      v += __shfl_xor(v, 8);   // masks < 16 stay within each 16-lane half (wave32)
      if (nl == 0 && row < Mrows) atomicAdd(&outAcc[bb * outStride + tt], v);
    }
    __syncthreads();
  }
}

// attention softmax over Tu + context; build x = [c_embed, context] and x2 low half
__global__ void att_ctx_kernel(const float* __restrict__ escore, const float* __restrict__ uenc,
                               const float* __restrict__ cemb,
                               float* __restrict__ ctx, float* __restrict__ xg,
                               float* __restrict__ x2) {
  int b = blockIdx.x, tid = threadIdx.x;   // 128 threads == Tu
  __shared__ float sm[128];
  __shared__ float att[128];
  float s = escore[b * TU_ + tid];
  sm[tid] = s; __syncthreads();
  for (int st = 64; st > 0; st >>= 1) { if (tid < st) sm[tid] = fmaxf(sm[tid], sm[tid + st]); __syncthreads(); }
  float mx = sm[0]; __syncthreads();
  float e = expf(s - mx);
  sm[tid] = e; __syncthreads();
  for (int st = 64; st > 0; st >>= 1) { if (tid < st) sm[tid] += sm[tid + st]; __syncthreads(); }
  float denom = sm[0];
  att[tid] = e / denom; __syncthreads();
  for (int hh = tid; hh < H_; hh += 128) {
    float acc = 0.f;
    for (int t = 0; t < TU_; ++t) acc += att[t] * uenc[(size_t)(t * B_ + b) * H_ + hh];
    ctx[b * H_ + hh] = acc;
    xg[b * 1024 + 512 + hh] = acc;   // x = [c_embed, context]
    x2[b * 1024 + hh] = acc;         // x2 = [context, h_new]
    xg[b * 1024 + hh] = cemb[b * H_ + hh];
  }
}

// GRU cell -> h_new ; fill x2 high half
__global__ void gru_kernel(const float* __restrict__ xg, const float* __restrict__ Wih,
                           const float* __restrict__ bih, const float* __restrict__ gh,
                           const float* __restrict__ h, float* __restrict__ hnew,
                           float* __restrict__ x2) {
  int idx = blockIdx.x * blockDim.x + threadIdx.x;
  if (idx >= B_ * H_) return;
  int b = idx >> 9, j = idx & 511;
  const float* xb = xg + b * 1024;
  const float* wr = Wih + (size_t)j * 1024;
  const float* wz = Wih + (size_t)(512 + j) * 1024;
  const float* wn = Wih + (size_t)(1024 + j) * 1024;
  float ar = bih[j], az = bih[512 + j], an = bih[1024 + j];
  for (int k = 0; k < 1024; ++k) {
    float xv = xb[k];
    ar += xv * wr[k]; az += xv * wz[k]; an += xv * wn[k];
  }
  float ghr = gh[b * 1536 + j], ghz = gh[b * 1536 + 512 + j], ghn = gh[b * 1536 + 1024 + j];
  float r = 1.f / (1.f + expf(-(ar + ghr)));
  float z = 1.f / (1.f + expf(-(az + ghz)));
  float n = tanhf(an + r * ghn);
  float hv = (1.f - z) * n + z * h[b * H_ + j];
  hnew[b * H_ + j] = hv;
  x2[b * 1024 + 512 + j] = hv;
}

__global__ void pgen_kernel(const float* __restrict__ ctx, const float* __restrict__ hnew,
                            const float* __restrict__ cemb, const float* __restrict__ pW,
                            const float* __restrict__ pb, float* __restrict__ pgen) {
  int b = blockIdx.x, tid = threadIdx.x;
  __shared__ float red[256];
  float acc = 0.f;
  for (int k = tid; k < 1536; k += 256) {
    float xv = (k < 512) ? ctx[b * 512 + k]
             : (k < 1024) ? hnew[b * 512 + k - 512]
             : cemb[b * 512 + k - 1024];
    acc += xv * pW[k];
  }
  red[tid] = acc; __syncthreads();
  for (int s = 128; s > 0; s >>= 1) { if (tid < s) red[tid] += red[tid + s]; __syncthreads(); }
  if (tid == 0) pgen[b] = 1.f / (1.f + expf(-(red[0] + pb[0])));
}

// gen_score = [context, h_new] @ proj_W.T + proj_b via WMMA.
// M=16 (8 real batch rows + 8 zero pad), N streamed over 32000, K=1024.
// proj_W (131 MB) is the dominant HBM traffic: one streaming pass, prefetched.
__global__ void proj_wmma_kernel(const float* __restrict__ x2,
                                 const float* __restrict__ projW,
                                 const float* __restrict__ projb,
                                 float* __restrict__ gen) {
  __shared__ _Float16 sX[16 * 1024];     // resident A operand (f16), [m][k]
  __shared__ _Float16 sB[8][32 * 16];    // per-wave B tile, K-major [k][n]
  int tid = threadIdx.x;                 // 256 threads = 8 waves
  int wave = tid >> 5, lane = tid & 31;
  for (int i = tid; i < 16 * 1024; i += 256) {
    int r = i >> 10;
    sX[i] = (r < B_) ? (_Float16)x2[i] : (_Float16)0.f;
  }
  __syncthreads();

  int n0 = (blockIdx.x * 8 + wave) * 16;
  v8f c = {0.f, 0.f, 0.f, 0.f, 0.f, 0.f, 0.f, 0.f};
  int rowB = lane & 15, halfk = lane >> 4;   // lane loads 16 k-contiguous floats of row n0+rowB
  for (int ks = 0; ks < 32; ++ks) {
    int k0 = ks * 32;
    const float* src = projW + (size_t)(n0 + rowB) * 1024 + k0 + halfk * 16;
    if (ks + 1 < 32) __builtin_prefetch(src + 32, 0, 3);   // global_prefetch next K chunk
#pragma unroll
    for (int i = 0; i < 16; ++i)
      sB[wave][(halfk * 16 + i) * 16 + rowB] = (_Float16)src[i];   // transpose to K-major
    __syncthreads();
    v16h a = ld_afrag(sX + k0, lane, 1024);
    v16h b = ld_bfrag(&sB[wave][0], lane, 16);
    c = __builtin_amdgcn_wmma_f32_16x16x32_f16(false, a, false, b, (short)0, c, false, false);
    __syncthreads();
  }
  int nl = lane & 15, half = lane >> 4;
#pragma unroll
  for (int r = 0; r < 8; ++r) {
    int m = r + 8 * half;
    if (m < B_) gen[(size_t)m * V_ + n0 + nl] = c[r] + projb[n0 + nl];
  }
}

// copy-score softmax stats + sparse scatter:
// u_copy contribution = EPS*sum(e) + scatter of e_t*(val-EPS) at one column per row.
__global__ void score_scatter_kernel(const float* __restrict__ score, const int* __restrict__ tok,
                                     float* __restrict__ mS, float* __restrict__ extra) {
  int b = blockIdx.x, tid = threadIdx.x;
  __shared__ float red[256];
  float lm = -1e30f;
  for (int t = tid; t < TH_; t += 256) lm = fmaxf(lm, score[b * TH_ + t]);
  red[tid] = lm; __syncthreads();
  for (int s = 128; s > 0; s >>= 1) { if (tid < s) red[tid] = fmaxf(red[tid], red[tid + s]); __syncthreads(); }
  float m = red[0]; __syncthreads();
  float ls = 0.f;
  for (int t = tid; t < TH_; t += 256) ls += expf(score[b * TH_ + t] - m);
  red[tid] = ls; __syncthreads();
  for (int s = 128; s > 0; s >>= 1) { if (tid < s) red[tid] += red[tid + s]; __syncthreads(); }
  if (tid == 0) { mS[b * 2] = m; mS[b * 2 + 1] = red[0]; }
  for (int t = 1 + tid; t < TH_; t += 256) {     // row 0 has no one-hot entry
    int w = tok[(t - 1) * B_ + b];
    float e = expf(score[b * TH_ + t] - m);
    bool cp = (w == 2) || (w >= V_);
    int col = cp ? (V_ + (t - 1)) : w;
    float val = cp ? 5.0f : 1.0f;
    atomicAdd(&extra[b * VT_ + col], e * (val - EPS_));
  }
}

// final fused softmax over [gen*pgen, u_copy*(1-pgen)] (64512) + proba assembly + h_new out
__global__ void final_kernel(const float* __restrict__ gen, const float* __restrict__ extra,
                             const float* __restrict__ mS, const float* __restrict__ pgen,
                             const float* __restrict__ hnew, float* __restrict__ out) {
  int b = blockIdx.x, tid = threadIdx.x;
  __shared__ float red[256];
  float pg = pgen[b];
  float m = mS[b * 2], S = mS[b * 2 + 1];
  float base = EPS_ * S;
  const float* gb = gen + (size_t)b * V_;
  const float* xb = extra + (size_t)b * VT_;
  const int NTOT = V_ + VT_;

  float lm = -1e30f;
  for (int i = tid; i < NTOT; i += 256) {
    float l = (i < V_) ? gb[i] * pg : (logf(base + xb[i - V_]) + m) * (1.f - pg);
    lm = fmaxf(lm, l);
  }
  red[tid] = lm; __syncthreads();
  for (int s = 128; s > 0; s >>= 1) { if (tid < s) red[tid] = fmaxf(red[tid], red[tid + s]); __syncthreads(); }
  float M = red[0]; __syncthreads();

  float ls = 0.f;
  for (int i = tid; i < NTOT; i += 256) {
    float l = (i < V_) ? gb[i] * pg : (logf(base + xb[i - V_]) + m) * (1.f - pg);
    ls += expf(l - M);
  }
  red[tid] = ls; __syncthreads();
  for (int s = 128; s > 0; s >>= 1) { if (tid < s) red[tid] += red[tid + s]; __syncthreads(); }
  float Z = red[0];

  for (int v = tid; v < VT_; v += 256) {
    float lu = (logf(base + xb[v]) + m) * (1.f - pg);
    float p = expf(lu - M) / Z;
    if (v < V_) p += expf(gb[v] * pg - M) / Z;
    out[(size_t)b * VT_ + v] = p;
  }
  for (int j = tid; j < H_; j += 256) {
    float hv = hnew[b * H_ + j];
    out[(size_t)B_ * VT_ + b * H_ + j] = hv;
    out[(size_t)B_ * VT_ + B_ * H_ + b * H_ + j] = hv;
  }
}

extern "C" void kernel_launch(void* const* d_in, const int* in_sizes, int n_in,
                              void* d_out, int out_size, void* d_ws, size_t ws_size,
                              hipStream_t stream) {
  (void)in_sizes; (void)n_in; (void)out_size; (void)ws_size;
  const int*   hist_tok = (const int*)  d_in[0];
  const float* hist_enc = (const float*)d_in[1];
  const float* u_enc    = (const float*)d_in[2];
  const int*   c_t      = (const int*)  d_in[3];
  const float* h0       = (const float*)d_in[4];
  const float* embW     = (const float*)d_in[5];
  const float* attnW    = (const float*)d_in[6];
  const float* attnb    = (const float*)d_in[7];
  const float* attnv    = (const float*)d_in[8];
  const float* Wih      = (const float*)d_in[9];
  const float* Whh      = (const float*)d_in[10];
  const float* bih      = (const float*)d_in[11];
  const float* bhh      = (const float*)d_in[12];
  const float* projW    = (const float*)d_in[13];
  const float* projb    = (const float*)d_in[14];
  const float* copyW    = (const float*)d_in[15];
  const float* copyb    = (const float*)d_in[16];
  const float* pgenW    = (const float*)d_in[17];
  const float* pgenb    = (const float*)d_in[18];
  float* ws  = (float*)d_ws;
  float* out = (float*)d_out;

  // workspace layout (floats)
  float* A1     = ws;                 // 8*512
  float* gh     = ws + 4096;          // 8*1536
  float* cemb   = ws + 16384;         // 8*512
  float* escore = ws + 20480;         // 8*128   (accum, zeroed)
  float* ctx    = ws + 21504;         // 8*512
  float* xg     = ws + 25600;         // 8*1024  [c_embed, context]
  float* x2     = ws + 33792;         // 8*1024  [context, h_new]
  float* hnew   = ws + 41984;         // 8*512
  float* gen    = ws + 46080;         // 8*32000
  float* score  = ws + 302080;        // 8*512   (accum, zeroed)
  float* pgen   = ws + 306176;        // 8
  float* mS     = ws + 306184;        // 8*2
  float* extra  = ws + 306200;        // 8*32512 (accum, zeroed)

  const size_t ldsGemm = (64 * 512 + 16 * 512) * sizeof(_Float16);  // 80 KB

  // zero accumulation buffers (deterministic every call)
  zero_kernel<<<4,    256, 0, stream>>>(escore, 8 * TU_);
  zero_kernel<<<16,   256, 0, stream>>>(score,  8 * TH_);
  zero_kernel<<<1016, 256, 0, stream>>>(extra,  8 * VT_);

  prep_kernel<<<80, 256, 0, stream>>>(h0, attnW, attnb, Whh, bhh, embW, c_t, A1, gh, cemb);

  // energy GEMM: rows = u_enc (Tu*B=1024), W2[j][k] = attn_W[j][512+k]
  gemm_reduce_kernel<<<dim3(8, 8), 128, ldsGemm, stream>>>(
      u_enc, attnW, 1024, 512, A1, attnv, escore, 1024, 8, TU_, 0);

  att_ctx_kernel<<<8, 128, 0, stream>>>(escore, u_enc, cemb, ctx, xg, x2);
  gru_kernel<<<16, 256, 0, stream>>>(xg, Wih, bih, gh, h0, hnew, x2);
  pgen_kernel<<<8, 256, 0, stream>>>(ctx, hnew, cemb, pgenW, pgenb, pgen);

  // projection GEMM over 32000 vocab rows (2000 N-tiles, 8 per block)
  proj_wmma_kernel<<<250, 256, 0, stream>>>(x2, projW, projb, gen);

  // copy GEMM: rows = hist_enc (Th*B=4096), W = copy2_W
  gemm_reduce_kernel<<<dim3(8, 16), 128, ldsGemm, stream>>>(
      hist_enc, copyW, 512, 0, copyb, hnew, score, 4096, 16, TH_, 1);

  score_scatter_kernel<<<8, 256, 0, stream>>>(score, hist_tok, mS, extra);
  final_kernel<<<8, 256, 0, stream>>>(gen, extra, mS, pgen, hnew, out);
}
